// globalAttention_29600914604685
// MI455X (gfx1250) — matrix operand order repaired and checked by reading, not compiled, not run on hardware
//
#include <hip/hip_runtime.h>
#include <hip/hip_bf16.h>
#include <stdint.h>

// ---------------------------------------------------------------------------
// MI455X (gfx1250) patch-attention block.
//  * dense convs + both attention GEMMs: v_wmma_f32_16x16x32_bf16 (f32 accum)
//  * attention GEMM tiles staged with the Tensor Data Mover
//    (tensor_load_to_lds + s_wait_tensorcnt), double-buffered in LDS
//  * q/k token-major, v d-major so every GEMM tile is a contiguous-row 2D
//    tile -> direct TDM descriptors, LDS rows padded via D#.pad (1DW/32DW)
//  * wave32, 256-thread blocks (8 waves)
// ---------------------------------------------------------------------------

typedef __attribute__((ext_vector_type(16))) __bf16 v16bf;
typedef __attribute__((ext_vector_type(8)))  float  v8f;
typedef __attribute__((ext_vector_type(4)))  unsigned int u32x4;
typedef __attribute__((ext_vector_type(8)))  int i32x8;
typedef __attribute__((ext_vector_type(4)))  int i32x4;

union Frag16 { uint32_t u[8]; v16bf v; };

__device__ __forceinline__ uint16_t f2bf(float f) {
    uint32_t u = __float_as_uint(f);
    u += 0x7FFFu + ((u >> 16) & 1u);          // round-to-nearest-even
    return (uint16_t)(u >> 16);
}

#define BB 16
#define TT 7
#define FF 112      // BB*TT frames
#define CC 64
#define SS 448      // TT * 64 tokens per batch
#define DD 4096     // CC * 64 features per token

// ---------------------------------------------------------------------------
// TDM: DMA a 64x64 bf16 tile (row stride `stride_elems`) from global memory
// into LDS at byte offset lds_addr, inserting 1 DWORD of padding after every
// 32 DWORDs (128B row -> 132B LDS pitch == uint16_t[66] rows).
// D# layout per CDNA5 ISA ch.8 (group0: count/lds/global/type, group1:
// data_size=2B, pad_enable, pad_interval=32DW, pad_amount=1DW, dims/strides).
// ---------------------------------------------------------------------------
__device__ __forceinline__ void tdm_load_tile64(uint32_t lds_addr,
                                                const uint16_t* gptr,
                                                uint32_t stride_elems) {
    uint64_t ga = (uint64_t)(uintptr_t)gptr;
    u32x4 g0;
    g0[0] = 1u;                                            // count=1, user mode
    g0[1] = lds_addr;                                      // LDS byte address
    g0[2] = (uint32_t)ga;                                  // global addr lo
    g0[3] = (uint32_t)((ga >> 32) & 0x01FFFFFFull)         // global addr hi
          | 0x80000000u;                                   // type=2 ("image")
    i32x8 g1;
    g1[0] = (int)((1u << 16) | (1u << 20) | (4u << 22));   // 2B elems, pad 1DW/32DW
    g1[1] = (int)(64u << 16);                              // tensor_dim0 = 64
    g1[2] = (int)(64u << 16);                              // tensor_dim1 = 64
    g1[3] = (int)(64u << 16);                              // tile_dim0 = 64
    g1[4] = 64;                                            // tile_dim1=64, tile_dim2=0
    g1[5] = (int)stride_elems;                             // tensor_dim0_stride lo
    g1[6] = 0;                                             // stride hi / dim1_stride
    g1[7] = 0;
    i32x4 z4 = {0, 0, 0, 0};
#if __has_include(<hip/amd_detail/amd_gfx1250_TDM.h>)
    i32x8 z8 = {0, 0, 0, 0, 0, 0, 0, 0};
    __builtin_amdgcn_tensor_load_to_lds(g0, g1, z4, z4, z8, 0);
#else
    __builtin_amdgcn_tensor_load_to_lds(g0, g1, z4, z4, 0);
#endif
}

// Shared WMMA micro-step for the attention GEMMs: one 64-wide K chunk,
// wave owns 1 M tile x 2 N tiles. Fragment k order: contiguous, k = 16*khalf + elem, same for A and B.
__device__ __forceinline__ void mma_step64(const uint16_t (*cA)[66],
                                           const uint16_t (*cB)[66],
                                           int mt, int ntb, int row16, int khalf,
                                           v8f& acc0, v8f& acc1) {
    #pragma unroll
    for (int kc = 0; kc < 2; ++kc) {
        Frag16 a;
        const uint16_t* arow = &cA[mt * 16 + row16][kc * 32 + khalf * 16];
        #pragma unroll
        for (int j = 0; j < 4; ++j) {          // A: K 16*khalf .. 16*khalf+15 contiguous
            a.u[j]     = *(const uint32_t*)(arow + 2 * j);
            a.u[j + 4] = *(const uint32_t*)(arow + 8 + 2 * j);
        }
        #pragma unroll
        for (int i = 0; i < 2; ++i) {
            Frag16 bm;                          // B: lane = N, elems = K in half
            const uint16_t* brow = &cB[(ntb + i) * 16 + row16][kc * 32 + khalf * 16];
            #pragma unroll
            for (int j = 0; j < 8; ++j)
                bm.u[j] = *(const uint32_t*)(brow + 2 * j);
            if (i == 0)
                acc0 = __builtin_amdgcn_wmma_f32_16x16x32_bf16(
                    false, a.v, false, bm.v, (short)0, acc0, false, false);
            else
                acc1 = __builtin_amdgcn_wmma_f32_16x16x32_bf16(
                    false, a.v, false, bm.v, (short)0, acc1, false, false);
        }
    }
}

// ---------------------------------------------------------------------------
// Kernel 1: fused depthwise 3x3 convs for q and k, + unfold8 + rearrange.
// Both q and k -> token-major bf16 (b, s, d).
// ---------------------------------------------------------------------------
__global__ __launch_bounds__(256) void dwconv_qk_kernel(
    const float* __restrict__ x,
    const float* __restrict__ wq, const float* __restrict__ bq,
    const float* __restrict__ wk, const float* __restrict__ bk,
    uint16_t* __restrict__ q_tok, uint16_t* __restrict__ k_tok)
{
    int gid = blockIdx.x * 256 + threadIdx.x;       // over FF*CC*64*64
    int xx = gid & 63;
    int yy = (gid >> 6) & 63;
    int c  = (gid >> 12) & 63;                      // uniform within a block
    int f  = gid >> 18;
    const float* xin = x + (((size_t)(f * 64 + c)) << 12);
    float qv = bq[c], kv = bk[c];
    #pragma unroll
    for (int dy = 0; dy < 3; ++dy) {
        int ys = yy + dy - 1;
        if ((unsigned)ys < 64u) {
            #pragma unroll
            for (int dx = 0; dx < 3; ++dx) {
                int xs = xx + dx - 1;
                if ((unsigned)xs < 64u) {
                    float iv = xin[ys * 64 + xs];
                    qv = fmaf(wq[c * 9 + dy * 3 + dx], iv, qv);
                    kv = fmaf(wk[c * 9 + dy * 3 + dx], iv, kv);
                }
            }
        }
    }
    int b = f / TT, t = f % TT;
    int d = c * 64 + (yy & 7) * 8 + (xx & 7);       // unfold8 channel index
    int s = t * 64 + (yy >> 3) * 8 + (xx >> 3);     // token index
    size_t tok = ((size_t)(b * SS + s)) << 12;
    q_tok[tok + d] = f2bf(qv);
    k_tok[tok + d] = f2bf(kv);
}

// ---------------------------------------------------------------------------
// Kernel 2/6: dense 3x3 conv as 9-tap WMMA GEMM.
// Block = (frame f, image row y): M = 64 out-ch, N = 64 pixels, K = 9x64.
// MODE 0: epilogue writes v d-major bf16 (b, d, s).
// MODE 1: epilogue adds bias + residual, writes fp32 (f,c,y,x) = d_out.
// ---------------------------------------------------------------------------
template <int MODE>
__global__ __launch_bounds__(256) void conv3x3_wmma_kernel(
    const float* __restrict__ in,       // (112,64,64,64) fp32
    const float* __restrict__ w,        // (64,64,3,3) OIHW
    const float* __restrict__ bias,     // (64)
    const float* __restrict__ resid,    // MODE 1: residual (same layout as out)
    float* __restrict__ out_f32,        // MODE 1 output
    uint16_t* __restrict__ out_dm)      // MODE 0 output: (16,4096,448) bf16
{
    __shared__ uint16_t sIn[3][66][66]; // [tap-row][x+1][in-ch], pad 66 halves
    __shared__ uint16_t sW[64][66];     // [oc][ic] for current tap

    const int tid = threadIdx.x;
    const int f = blockIdx.x >> 6;
    const int y = blockIdx.x & 63;

    // Stage input rows y-1..y+1, x=-1..64, all 64 channels (zero-padded) bf16.
    for (int idx = tid; idx < 3 * 64 * 66; idx += 256) {
        int col = idx % 66;                  // consecutive threads -> x (coalesced)
        int ic  = (idx / 66) & 63;
        int row = idx / (66 * 64);
        int ys = y + row - 1;
        int xs = col - 1;
        float v = 0.f;
        if ((unsigned)ys < 64u && (unsigned)xs < 64u)
            v = in[(((size_t)(f * 64 + ic)) << 12) + ys * 64 + xs];
        sIn[row][col][ic] = f2bf(v);
    }

    const int lane  = tid & 31;
    const int wv    = tid >> 5;
    const int row16 = lane & 15;
    const int khalf = lane >> 4;
    const int mt  = wv >> 1;             // wave's M tile (oc/16)
    const int ntb = (wv & 1) * 2;        // first of wave's two N tiles

    v8f acc0 = {}; v8f acc1 = {};

    for (int dy = 0; dy < 3; ++dy)
    for (int dx = 0; dx < 3; ++dx) {
        if (dy * 3 + dx < 8)             // prefetch next tap's weights (L2)
            __builtin_prefetch(w + (size_t)tid * 9 + dy * 3 + dx + 1, 0, 1);
        __syncthreads();
        for (int idx = tid; idx < 64 * 64; idx += 256) {
            int ic = idx & 63, oc = idx >> 6;
            sW[oc][ic] = f2bf(w[((oc * 64 + ic) * 3 + dy) * 3 + dx]);
        }
        __syncthreads();
        #pragma unroll
        for (int kc = 0; kc < 2; ++kc) {         // in-ch halves 0..31 / 32..63
            Frag16 a;                            // A: 16(oc) x 32(ic)
            const uint16_t* arow = &sW[mt * 16 + row16][kc * 32 + khalf * 16];
            #pragma unroll
            for (int j = 0; j < 4; ++j) {
                a.u[j]     = *(const uint32_t*)(arow + 2 * j);
                a.u[j + 4] = *(const uint32_t*)(arow + 8 + 2 * j);
            }
            #pragma unroll
            for (int i = 0; i < 2; ++i) {
                int nt = ntb + i;
                Frag16 bm;                       // B: 32(ic) x 16(x); lane = N
                const uint16_t* brow = &sIn[dy][nt * 16 + row16 + dx][kc * 32 + khalf * 16];
                #pragma unroll
                for (int j = 0; j < 8; ++j)
                    bm.u[j] = *(const uint32_t*)(brow + 2 * j);
                if (i == 0)
                    acc0 = __builtin_amdgcn_wmma_f32_16x16x32_bf16(
                        false, a.v, false, bm.v, (short)0, acc0, false, false);
                else
                    acc1 = __builtin_amdgcn_wmma_f32_16x16x32_bf16(
                        false, a.v, false, bm.v, (short)0, acc1, false, false);
            }
        }
    }

    const int bB = f / TT, t = f % TT;
    const int ki = y & 7, nh = y >> 3;
    #pragma unroll
    for (int i = 0; i < 2; ++i) {
        int nt = ntb + i;
        #pragma unroll
        for (int r = 0; r < 8; ++r) {
            int oc = mt * 16 + r + khalf * 8;    // ISA C/D layout
            int xg = nt * 16 + row16;
            float val = (i == 0) ? acc0[r] : acc1[r];
            if (MODE == 1) {
                size_t addr = ((size_t)(f * 64 + oc) << 12) + y * 64 + xg;
                out_f32[addr] = val + bias[oc] + resid[addr];
            } else {
                int d = oc * 64 + ki * 8 + (xg & 7);
                int s = t * 64 + nh * 8 + (xg >> 3);
                out_dm[((size_t)(bB * DD + d)) * SS + s] = f2bf(val + bias[oc]);
            }
        }
    }
}

// ---------------------------------------------------------------------------
// Kernel 3: attention logits attn[b][s][u] = (1/64) * sum_d q[s][d]*k[u][d].
// 64x64 tile per block; K=4096 in 64-wide steps; A/B tiles DMA'd by the TDM
// into double-buffered LDS (wave 0 drives the TDM, TENSORcnt pipelines it).
// ---------------------------------------------------------------------------
__global__ __launch_bounds__(256) void gemm_qk_kernel(
    const uint16_t* __restrict__ q_tok,
    const uint16_t* __restrict__ k_tok,
    float* __restrict__ attnL)
{
    __shared__ uint16_t sA[2][64][66];
    __shared__ uint16_t sB[2][64][66];
    const int tid = threadIdx.x, lane = tid & 31, wv = tid >> 5;
    const int b  = blockIdx.y;
    const int m0 = (blockIdx.x / 7) * 64;
    const int n0 = (blockIdx.x % 7) * 64;
    const int row16 = lane & 15, khalf = lane >> 4;
    const int mt = wv >> 1, ntb = (wv & 1) * 2;

    const uint16_t* qb = q_tok + (((size_t)(b * SS + m0)) << 12);
    const uint16_t* kb = k_tok + (((size_t)(b * SS + n0)) << 12);
    const uint32_t ldsA[2] = { (uint32_t)(uintptr_t)&sA[0][0][0],
                               (uint32_t)(uintptr_t)&sA[1][0][0] };
    const uint32_t ldsB[2] = { (uint32_t)(uintptr_t)&sB[0][0][0],
                               (uint32_t)(uintptr_t)&sB[1][0][0] };

    v8f acc0 = {}; v8f acc1 = {};

    if (wv == 0) {                                // prologue: fill buffer 0
        tdm_load_tile64(ldsA[0], qb, DD);
        tdm_load_tile64(ldsB[0], kb, DD);
    }
    for (int ks = 0; ks < 64; ++ks) {
        __syncthreads();                          // next buffer free to overwrite
        if (wv == 0) {
            if (ks + 1 < 64) {
                int nb = (ks + 1) & 1;
                tdm_load_tile64(ldsA[nb], qb + (ks + 1) * 64, DD);
                tdm_load_tile64(ldsB[nb], kb + (ks + 1) * 64, DD);
                __builtin_amdgcn_s_wait_tensorcnt(2);   // current buffer done
            } else {
                __builtin_amdgcn_s_wait_tensorcnt(0);
            }
        }
        __syncthreads();                          // current buffer visible to all
        mma_step64(sA[ks & 1], sB[ks & 1], mt, ntb, row16, khalf, acc0, acc1);
    }
    #pragma unroll
    for (int i = 0; i < 2; ++i) {
        #pragma unroll
        for (int r = 0; r < 8; ++r) {
            int m = m0 + mt * 16 + r + khalf * 8;
            int n = n0 + (ntb + i) * 16 + row16;
            float val = ((i == 0) ? acc0[r] : acc1[r]) * 0.015625f; // 1/sqrt(4096)
            attnL[((size_t)(b * SS + m)) * SS + n] = val;
        }
    }
}

// ---------------------------------------------------------------------------
// Kernel 4: row softmax over 448 columns; one wave per row (14 elems/lane).
// ---------------------------------------------------------------------------
__global__ __launch_bounds__(256) void softmax_kernel(
    const float* __restrict__ attnL, uint16_t* __restrict__ attn_bf)
{
    const int lane = threadIdx.x & 31, wv = threadIdx.x >> 5;
    const int rid = blockIdx.x * 8 + wv;          // 0 .. 16*448-1
    const float* rowp = attnL + (size_t)rid * SS;
    float v[14];
    float mx = -3.0e38f;
    #pragma unroll
    for (int j = 0; j < 14; ++j) { v[j] = rowp[lane + j * 32]; mx = fmaxf(mx, v[j]); }
    #pragma unroll
    for (int o = 16; o > 0; o >>= 1) mx = fmaxf(mx, __shfl_xor(mx, o, 32));
    float sum = 0.f;
    #pragma unroll
    for (int j = 0; j < 14; ++j) { v[j] = __expf(v[j] - mx); sum += v[j]; }
    #pragma unroll
    for (int o = 16; o > 0; o >>= 1) sum += __shfl_xor(sum, o, 32);
    float inv = 1.0f / sum;
    uint16_t* orow = attn_bf + (size_t)rid * SS;
    #pragma unroll
    for (int j = 0; j < 14; ++j) orow[lane + j * 32] = f2bf(v[j] * inv);
}

// ---------------------------------------------------------------------------
// Kernel 5: out[s][d] = sum_u attn[s][u] * v[d][u], fold8 scatter to feat.
// A = probs (M=448 x K=448, row-major), B = v d-major -> both tiles are
// contiguous-row 2D tiles, DMA'd by the TDM (double-buffered).
// ---------------------------------------------------------------------------
__global__ __launch_bounds__(256) void gemm_av_kernel(
    const uint16_t* __restrict__ attn_bf,
    const uint16_t* __restrict__ v_dm,
    float* __restrict__ feat)
{
    __shared__ uint16_t sA[2][64][66];
    __shared__ uint16_t sB[2][64][66];
    const int tid = threadIdx.x, lane = tid & 31, wv = tid >> 5;
    const int b  = blockIdx.z;
    const int t  = blockIdx.y;        // m0 = t*64 (token tile == one frame)
    const int c  = blockIdx.x;        // n0 = c*64 (d tile == one channel)
    const int m0 = t * 64, n0 = c * 64;
    const int row16 = lane & 15, khalf = lane >> 4;
    const int mt = wv >> 1, ntb = (wv & 1) * 2;

    const uint16_t* ab = attn_bf + (size_t)(b * SS + m0) * SS;   // rows: stride 448
    const uint16_t* vb = v_dm + ((size_t)(b * DD + n0)) * SS;    // rows: stride 448
    const uint32_t ldsA[2] = { (uint32_t)(uintptr_t)&sA[0][0][0],
                               (uint32_t)(uintptr_t)&sA[1][0][0] };
    const uint32_t ldsB[2] = { (uint32_t)(uintptr_t)&sB[0][0][0],
                               (uint32_t)(uintptr_t)&sB[1][0][0] };

    v8f acc0 = {}; v8f acc1 = {};

    if (wv == 0) {
        tdm_load_tile64(ldsA[0], ab, SS);
        tdm_load_tile64(ldsB[0], vb, SS);
    }
    for (int ks = 0; ks < 7; ++ks) {
        __syncthreads();
        if (wv == 0) {
            if (ks + 1 < 7) {
                int nb = (ks + 1) & 1;
                tdm_load_tile64(ldsA[nb], ab + (ks + 1) * 64, SS);
                tdm_load_tile64(ldsB[nb], vb + (ks + 1) * 64, SS);
                __builtin_amdgcn_s_wait_tensorcnt(2);
            } else {
                __builtin_amdgcn_s_wait_tensorcnt(0);
            }
        }
        __syncthreads();
        mma_step64(sA[ks & 1], sB[ks & 1], mt, ntb, row16, khalf, acc0, acc1);
    }
    const int f = b * TT + t;
    #pragma unroll
    for (int i = 0; i < 2; ++i) {
        #pragma unroll
        for (int r = 0; r < 8; ++r) {
            int l    = mt * 16 + r + khalf * 8;       // token within frame
            int dloc = (ntb + i) * 16 + row16;        // d within channel c
            int yy = (l >> 3) * 8 + (dloc >> 3);      // nh*8 + ki
            int xx = (l & 7) * 8 + (dloc & 7);        // nw*8 + kj
            float val = (i == 0) ? acc0[r] : acc1[r];
            feat[(((size_t)(f * 64 + c)) << 12) + yy * 64 + xx] = val;
        }
    }
}

// ---------------------------------------------------------------------------
// Host launcher
// ---------------------------------------------------------------------------
extern "C" void kernel_launch(void* const* d_in, const int* in_sizes, int n_in,
                              void* d_out, int out_size, void* d_ws, size_t ws_size,
                              hipStream_t stream) {
    (void)in_sizes; (void)n_in; (void)out_size; (void)ws_size;
    const float* x  = (const float*)d_in[0];
    const float* wq = (const float*)d_in[1];
    const float* bq = (const float*)d_in[2];
    const float* wk = (const float*)d_in[3];
    const float* bk = (const float*)d_in[4];
    const float* wv = (const float*)d_in[5];
    const float* bv = (const float*)d_in[6];
    const float* wc = (const float*)d_in[7];
    const float* bc = (const float*)d_in[8];

    char* ws = (char*)d_ws;
    const size_t QS = (size_t)BB * SS * DD * 2;             // 58,720,256 B
    uint16_t* q_tok  = (uint16_t*)(ws);
    uint16_t* k_tok  = (uint16_t*)(ws + QS);
    uint16_t* v_dm   = (uint16_t*)(ws + 2 * QS);
    float*    attnL  = (float*)   (ws + 3 * QS);            // 16*448*448 f32
    uint16_t* attnbf = (uint16_t*)(ws + 3 * QS + (size_t)BB * SS * SS * 4);
    float*    feat   = (float*)   (ws + 3 * QS + (size_t)BB * SS * SS * 6);

    // 1) depthwise q/k convs + layout transform (token-major bf16)
    dwconv_qk_kernel<<<(FF * CC * 64 * 64) / 256, 256, 0, stream>>>(
        x, wq, bq, wk, bk, q_tok, k_tok);
    // 2) dense v conv (WMMA) -> d-major bf16
    conv3x3_wmma_kernel<0><<<FF * 64, 256, 0, stream>>>(
        x, wv, bv, nullptr, nullptr, v_dm);
    // 3) Q^T K logits (WMMA + TDM staging)
    gemm_qk_kernel<<<dim3(49, BB), 256, 0, stream>>>(q_tok, k_tok, attnL);
    // 4) softmax -> bf16 probs
    softmax_kernel<<<(BB * SS) / 8, 256, 0, stream>>>(attnL, attnbf);
    // 5) Attn * V (WMMA + TDM staging) + fold8 -> feat fp32
    gemm_av_kernel<<<dim3(64, TT, BB), 256, 0, stream>>>(attnbf, v_dm, feat);
    // 6) final dense conv (WMMA) + bias + residual -> d_out
    conv3x3_wmma_kernel<1><<<FF * 64, 256, 0, stream>>>(
        feat, wc, bc, x, (float*)d_out, nullptr);
}
